// SimpleALIGNNLike_54554674594125
// MI455X (gfx1250) — matrix operand order repaired and checked by reading, not compiled, not run on hardware
//
#include <hip/hip_runtime.h>
#include <hip/hip_bf16.h>
#include <hip/hip_fp16.h>

#define NUM_GRAPHS 512
#define MAX_NODES  192
#define TOTAL_N    65536
#define DMODEL     256
#define NHEAD      4
#define DHEAD      64
#define QKVLD      768
#define GK         256
#define BM         256
#define BN         64

typedef __attribute__((ext_vector_type(16))) _Float16     v16h;
typedef __attribute__((ext_vector_type(8)))  float        v8f;
typedef __attribute__((ext_vector_type(4)))  unsigned int u32x4;

union HF16 { v16h v; _Float16 h[16]; unsigned int u[8]; u32x4 q[2]; };

__device__ __forceinline__ v8f wmma_f32_f16(v16h a, v16h b, v8f c) {
  // D = A(16x32 f16) * B(32x16 f16) + C(16x16 f32)
  return __builtin_amdgcn_wmma_f32_16x16x32_f16(false, a, false, b, (short)0, c, false, false);
}

// A fragment 16x32: element A[m][k] = base[(row0+m)*ld + k0 + k]
// ISA layout: lane(0-15)->M, VGPR0-3: K=(0..7)+8*hi, VGPR4-7: K=(16..23)+8*hi
// Two 16-byte contiguous groups -> ds_load_b128 x2 (all call sites 16B aligned).
__device__ __forceinline__ v16h load_fragA(const _Float16* base, int row0, int ld, int k0) {
  int l = threadIdx.x & 31;
  int m = l & 15, hi = l >> 4;
  const _Float16* p = base + (row0 + m) * ld + k0 + 8 * hi;
  HF16 f;
  f.q[0] = *(const u32x4*)p;
  f.q[1] = *(const u32x4*)(p + 16);
  return f.v;
}

// B fragment 32x16 from row-major "B-transposed" storage: B[k][n] = base[(row0+n)*ld + k0 + k]
// ISA layout: lane(0-15)->N, VGPR0-3: K=(0..7)+8*hi, VGPR4-7: K=(16..23)+8*hi
// Two 16-byte contiguous groups -> ds_load_b128 x2.
__device__ __forceinline__ v16h load_fragB_rm(const _Float16* base, int row0, int ld, int k0) {
  int l = threadIdx.x & 31;
  int n = l & 15, hi = l >> 4;
  const _Float16* p = base + (row0 + n) * ld + k0 + 8 * hi;
  HF16 f;
  f.q[0] = *(const u32x4*)p;
  f.q[1] = *(const u32x4*)(p + 16);
  return f.v;
}

// ---------------------------------------------------------------- utilities
__global__ void starts_kernel(const int* __restrict__ batch, int* __restrict__ starts) {
  int g = blockIdx.x * blockDim.x + threadIdx.x;
  if (g >= NUM_GRAPHS) return;
  int lo = 0, hi = TOTAL_N;               // lower_bound of g (batch is sorted)
  while (lo < hi) { int mid = (lo + hi) >> 1; if (batch[mid] < g) lo = mid + 1; else hi = mid; }
  starts[g] = lo;
  if (g == NUM_GRAPHS - 1) starts[NUM_GRAPHS] = TOTAL_N;
}

__global__ void cast_f32_f16_kernel(const float* __restrict__ in, _Float16* __restrict__ out, int nelem) {
  int i = blockIdx.x * 256 + threadIdx.x;
  if (i < nelem) out[i] = (_Float16)in[i];
}

// ------------------------------------------------- generic GEMM  C = A*W^T+b
// A:[M,256] f16, W:[N,256] f16, C:[M,ldc] f16.  256x64 tile, 256 threads,
// each wave computes a 32x64 tile; k-loop software-pipelined.
__global__ __launch_bounds__(256) void gemm_xwT_kernel(
    const _Float16* __restrict__ A, const _Float16* __restrict__ W,
    const float* __restrict__ bias, _Float16* __restrict__ C, int ldc, int M, int N) {
  extern __shared__ _Float16 lds[];
  _Float16* As = lds;            // 256 x 256
  _Float16* Ws = lds + BM * GK;  // 64 x 256
  const int t = threadIdx.x;
  const int m0 = blockIdx.x * BM;
  const int n0 = blockIdx.y * BN;

  const unsigned int* A32 = (const unsigned int*)A;
  const unsigned int* W32 = (const unsigned int*)W;
  unsigned int* As32 = (unsigned int*)As;
  unsigned int* Ws32 = (unsigned int*)Ws;
#pragma unroll 4
  for (int i = 0; i < (BM * GK / 2) / 256; ++i) {   // 256 rows * 128 dwords
    int li = i * 256 + t;
    int row = m0 + (li >> 7);
    As32[li] = (row < M) ? A32[(size_t)row * (GK / 2) + (li & 127)] : 0u;
  }
#pragma unroll 4
  for (int i = 0; i < (BN * GK / 2) / 256; ++i) {   // 64 rows * 128 dwords
    int li = i * 256 + t;
    int row = n0 + (li >> 7);
    Ws32[li] = (row < N) ? W32[(size_t)row * (GK / 2) + (li & 127)] : 0u;
  }
  __syncthreads();

  const int w = t >> 5;                    // wave -> 32-row block
  const int r0 = w * 32;
  v8f acc00 = {}, acc01 = {}, acc02 = {}, acc03 = {};
  v8f acc10 = {}, acc11 = {}, acc12 = {}, acc13 = {};

  v16h a0 = load_fragA(As, r0, GK, 0);
  v16h a1 = load_fragA(As, r0 + 16, GK, 0);
  v16h b0 = load_fragB_rm(Ws, 0, GK, 0);
  v16h b1 = load_fragB_rm(Ws, 16, GK, 0);
  v16h b2 = load_fragB_rm(Ws, 32, GK, 0);
  v16h b3 = load_fragB_rm(Ws, 48, GK, 0);
#pragma unroll
  for (int ks = 0; ks < 8; ++ks) {
    v16h a0n, a1n, b0n, b1n, b2n, b3n;
    if (ks < 7) {                      // prefetch next k-step while computing
      int k = (ks + 1) * 32;
      a0n = load_fragA(As, r0, GK, k);
      a1n = load_fragA(As, r0 + 16, GK, k);
      b0n = load_fragB_rm(Ws, 0, GK, k);
      b1n = load_fragB_rm(Ws, 16, GK, k);
      b2n = load_fragB_rm(Ws, 32, GK, k);
      b3n = load_fragB_rm(Ws, 48, GK, k);
    }
    acc00 = wmma_f32_f16(a0, b0, acc00);
    acc10 = wmma_f32_f16(a1, b0, acc10);
    acc01 = wmma_f32_f16(a0, b1, acc01);
    acc11 = wmma_f32_f16(a1, b1, acc11);
    acc02 = wmma_f32_f16(a0, b2, acc02);
    acc12 = wmma_f32_f16(a1, b2, acc12);
    acc03 = wmma_f32_f16(a0, b3, acc03);
    acc13 = wmma_f32_f16(a1, b3, acc13);
    if (ks < 7) { a0 = a0n; a1 = a1n; b0 = b0n; b1 = b1n; b2 = b2n; b3 = b3n; }
  }

  const int l = t & 31, n = l & 15, hi = l >> 4;
  v8f accs0[4] = { acc00, acc01, acc02, acc03 };
  v8f accs1[4] = { acc10, acc11, acc12, acc13 };
#pragma unroll
  for (int c = 0; c < 4; ++c) {
    int gn = n0 + c * 16 + n;
    float bv = (gn < N) ? bias[gn] : 0.f;
#pragma unroll
    for (int j = 0; j < 8; ++j) {
      int gm = m0 + r0 + j + 8 * hi;
      if (gm < M && gn < N) C[(size_t)gm * ldc + gn] = (_Float16)(accs0[c][j] + bv);
      int gm1 = gm + 16;
      if (gm1 < M && gn < N) C[(size_t)gm1 * ldc + gn] = (_Float16)(accs1[c][j] + bv);
    }
  }
}

// ------------------------------------------------- per (graph, head) attention
__global__ __launch_bounds__(256) void attn_kernel(
    const _Float16* __restrict__ QKV, const int* __restrict__ starts,
    _Float16* __restrict__ O) {
  extern __shared__ _Float16 smem[];
  _Float16* Qs  = smem;                         // 192 x 64 (pre-scaled 1/8)
  _Float16* Ks  = Qs + MAX_NODES * DHEAD;       // 192 x 64
  _Float16* VsT = Ks + MAX_NODES * DHEAD;       // 64 x 192  (V transposed)
  _Float16* Ps  = VsT + MAX_NODES * DHEAD;      // 8 waves x 16 x 192

  const int g = blockIdx.x, h = blockIdx.y;
  const int s0 = starts[g];
  const int L  = starts[g + 1] - s0;
  const int t = threadIdx.x, w = t >> 5, l = t & 31;
  const int n = l & 15, hi = l >> 4;

  for (int i = 0; i < (MAX_NODES * DHEAD) / 256; ++i) {
    int li = i * 256 + t;
    int r = li >> 6, col = li & 63;
    _Float16 qv = (_Float16)0.f, kv = (_Float16)0.f, vv = (_Float16)0.f;
    if (r < L) {
      const _Float16* row = QKV + (size_t)(s0 + r) * QKVLD + h * DHEAD + col;
      qv = (_Float16)((float)row[0] * 0.125f);   // 1/sqrt(DH)
      kv = row[DMODEL];
      vv = row[2 * DMODEL];
    }
    Qs[li] = qv;
    Ks[li] = kv;
    VsT[col * MAX_NODES + r] = vv;               // transpose V: [DHEAD][MAX_NODES]
  }
  __syncthreads();

  const int nQ  = (L + 15) >> 4;
  const int nK  = (L + 15) >> 4;
  const int nks = (L + 31) >> 5;
  _Float16* Pw = Ps + w * (16 * MAX_NODES);

  for (int rb = w; rb < nQ; rb += 8) {
    v16h a0 = load_fragA(Qs, rb * 16, DHEAD, 0);
    v16h a1 = load_fragA(Qs, rb * 16, DHEAD, 32);
    v8f acc[12] = {};
#pragma unroll
    for (int c = 0; c < 12; ++c) {
      if (c < nK) {
        v16h b0 = load_fragB_rm(Ks, c * 16, DHEAD, 0);
        v16h b1 = load_fragB_rm(Ks, c * 16, DHEAD, 32);
        v8f s = acc[c];
        s = wmma_f32_f16(a0, b0, s);
        s = wmma_f32_f16(a1, b1, s);
        acc[c] = s;
      }
    }
    // mask invalid keys, row-max over the 16 columns held by each half-wave
    float rmax[8];
#pragma unroll
    for (int j = 0; j < 8; ++j) rmax[j] = -1e30f;
#pragma unroll
    for (int c = 0; c < 12; ++c) {
      if (c < nK) {
        bool oob = (c * 16 + n) >= L;
#pragma unroll
        for (int j = 0; j < 8; ++j) {
          float sv = oob ? -1e30f : acc[c][j];
          acc[c][j] = sv;
          rmax[j] = fmaxf(rmax[j], sv);
        }
      }
    }
#pragma unroll
    for (int d = 1; d <= 8; d <<= 1)
#pragma unroll
      for (int j = 0; j < 8; ++j) rmax[j] = fmaxf(rmax[j], __shfl_xor(rmax[j], d, 32));
    float rsum[8];
#pragma unroll
    for (int j = 0; j < 8; ++j) rsum[j] = 0.f;
#pragma unroll
    for (int c = 0; c < 12; ++c) {
      if (c < nK) {
#pragma unroll
        for (int j = 0; j < 8; ++j) {
          float p = __expf(acc[c][j] - rmax[j]);   // exp(-huge)=0 for masked
          acc[c][j] = p;
          rsum[j] += p;
        }
      }
    }
#pragma unroll
    for (int d = 1; d <= 8; d <<= 1)
#pragma unroll
      for (int j = 0; j < 8; ++j) rsum[j] += __shfl_xor(rsum[j], d, 32);
    float rinv[8];
#pragma unroll
    for (int j = 0; j < 8; ++j) rinv[j] = 1.f / rsum[j];
    // normalized P -> per-wave LDS (zero-padded to 192 cols for the k-loop)
#pragma unroll
    for (int c = 0; c < 12; ++c) {
#pragma unroll
      for (int j = 0; j < 8; ++j) {
        int m = j + 8 * hi;
        float p = (c < nK) ? acc[c][j] * rinv[j] : 0.f;
        Pw[m * MAX_NODES + c * 16 + n] = (_Float16)p;
      }
    }
    // O = P @ V  (B[k][n] = V[k][n] = VsT[n*192 + k] -> row-major B loads)
    v8f oacc[4] = {};
#pragma unroll
    for (int ks = 0; ks < 6; ++ks) {
      if (ks < nks) {
        v16h pa = load_fragA(Pw, 0, MAX_NODES, ks * 32);
#pragma unroll
        for (int cb = 0; cb < 4; ++cb) {
          v16h vb = load_fragB_rm(VsT, cb * 16, MAX_NODES, ks * 32);
          oacc[cb] = wmma_f32_f16(pa, vb, oacc[cb]);
        }
      }
    }
#pragma unroll
    for (int cb = 0; cb < 4; ++cb) {
#pragma unroll
      for (int j = 0; j < 8; ++j) {
        int m = j + 8 * hi;
        int r = rb * 16 + m;
        if (r < L)
          O[(size_t)(s0 + r) * DMODEL + h * DHEAD + cb * 16 + n] = (_Float16)oacc[cb][j];
      }
    }
  }
}

// ------------------------------------------------- masked mean-pool + MLP
__global__ __launch_bounds__(256) void pool_readout_kernel(
    const _Float16* __restrict__ Hfin, const int* __restrict__ starts,
    const float* __restrict__ rw1, const float* __restrict__ rb1,
    const float* __restrict__ rw2, const float* __restrict__ rb2,
    const float* __restrict__ rw3, const float* __restrict__ rb3,
    float* __restrict__ out) {
  __shared__ float pooled[DMODEL];
  __shared__ float h1s[DMODEL];
  __shared__ float h2s[128];
  const int g = blockIdx.x, t = threadIdx.x;
  const int s0 = starts[g];
  const int L = starts[g + 1] - s0;
  float acc = 0.f;
  for (int r = 0; r < L; ++r) acc += (float)Hfin[(size_t)(s0 + r) * DMODEL + t];
  pooled[t] = acc / (float)L;
  __syncthreads();
  float d1 = rb1[t];
  for (int k = 0; k < DMODEL; ++k) d1 += rw1[t * DMODEL + k] * pooled[k];
  h1s[t] = fmaxf(d1, 0.f);
  __syncthreads();
  if (t < 128) {
    float d2 = rb2[t];
    for (int k = 0; k < DMODEL; ++k) d2 += rw2[t * DMODEL + k] * h1s[k];
    h2s[t] = fmaxf(d2, 0.f);
  }
  __syncthreads();
  if (t == 0) {
    float e = rb3[0];
    for (int k = 0; k < 128; ++k) e += rw3[k] * h2s[k];
    out[g] = e;
  }
}

// ---------------------------------------------------------------- launcher
extern "C" void kernel_launch(void* const* d_in, const int* in_sizes, int n_in,
                              void* d_out, int out_size, void* d_ws, size_t ws_size,
                              hipStream_t stream) {
  (void)in_sizes; (void)n_in; (void)out_size; (void)ws_size;
  const float* x      = (const float*)d_in[0];
  const int*   batch  = (const int*)d_in[1];
  const float* in_w1  = (const float*)d_in[2];
  const float* in_b1  = (const float*)d_in[3];
  const float* out_w1 = (const float*)d_in[4];
  const float* out_b1 = (const float*)d_in[5];
  const float* in_w2  = (const float*)d_in[6];
  const float* in_b2  = (const float*)d_in[7];
  const float* out_w2 = (const float*)d_in[8];
  const float* out_b2 = (const float*)d_in[9];
  const float* r_w1   = (const float*)d_in[10];
  const float* r_b1   = (const float*)d_in[11];
  const float* r_w2   = (const float*)d_in[12];
  const float* r_b2   = (const float*)d_in[13];
  const float* r_w3   = (const float*)d_in[14];
  const float* r_b3   = (const float*)d_in[15];
  float* out = (float*)d_out;

  size_t off = 0;
  auto carve = [&](size_t bytes) -> void* {
    size_t a = (off + 255) & ~(size_t)255;
    off = a + bytes;
    return (void*)((char*)d_ws + a);
  };
  int*      starts = (int*)carve((NUM_GRAPHS + 1) * sizeof(int));
  _Float16* Abuf = (_Float16*)carve((size_t)TOTAL_N * DMODEL * 2);  // Xh -> H1 -> H2
  _Float16* QKV  = (_Float16*)carve((size_t)TOTAL_N * QKVLD * 2);   // layer 1 then 2
  _Float16* Obuf = (_Float16*)carve((size_t)TOTAL_N * DMODEL * 2);  // attn out 1 then 2
  _Float16* w1h  = (_Float16*)carve((size_t)3 * DMODEL * DMODEL * 2);
  _Float16* ow1h = (_Float16*)carve((size_t)DMODEL * DMODEL * 2);
  _Float16* w2h  = (_Float16*)carve((size_t)3 * DMODEL * DMODEL * 2);
  _Float16* ow2h = (_Float16*)carve((size_t)DMODEL * DMODEL * 2);

  starts_kernel<<<2, 256, 0, stream>>>(batch, starts);
  cast_f32_f16_kernel<<<(TOTAL_N * DMODEL) / 256, 256, 0, stream>>>(x, Abuf, TOTAL_N * DMODEL);
  cast_f32_f16_kernel<<<(3 * DMODEL * DMODEL) / 256, 256, 0, stream>>>(in_w1, w1h, 3 * DMODEL * DMODEL);
  cast_f32_f16_kernel<<<(DMODEL * DMODEL) / 256, 256, 0, stream>>>(out_w1, ow1h, DMODEL * DMODEL);
  cast_f32_f16_kernel<<<(3 * DMODEL * DMODEL) / 256, 256, 0, stream>>>(in_w2, w2h, 3 * DMODEL * DMODEL);
  cast_f32_f16_kernel<<<(DMODEL * DMODEL) / 256, 256, 0, stream>>>(out_w2, ow2h, DMODEL * DMODEL);

  dim3 gq(TOTAL_N / BM, QKVLD / BN);
  dim3 go(TOTAL_N / BM, DMODEL / BN);
  size_t gemmLds = (size_t)(BM * GK + BN * GK) * sizeof(_Float16);              // 160 KB
  size_t attnLds = (size_t)(3 * MAX_NODES * DHEAD + 8 * 16 * MAX_NODES) * 2;    // 120 KB

  // layer 1
  gemm_xwT_kernel<<<gq, 256, gemmLds, stream>>>(Abuf, w1h, in_b1, QKV, QKVLD, TOTAL_N, QKVLD);
  attn_kernel<<<dim3(NUM_GRAPHS, NHEAD), 256, attnLds, stream>>>(QKV, starts, Obuf);
  gemm_xwT_kernel<<<go, 256, gemmLds, stream>>>(Obuf, ow1h, out_b1, Abuf, DMODEL, TOTAL_N, DMODEL);
  // layer 2
  gemm_xwT_kernel<<<gq, 256, gemmLds, stream>>>(Abuf, w2h, in_b2, QKV, QKVLD, TOTAL_N, QKVLD);
  attn_kernel<<<dim3(NUM_GRAPHS, NHEAD), 256, attnLds, stream>>>(QKV, starts, Obuf);
  gemm_xwT_kernel<<<go, 256, gemmLds, stream>>>(Obuf, ow2h, out_b2, Abuf, DMODEL, TOTAL_N, DMODEL);
  // pool + readout
  pool_readout_kernel<<<NUM_GRAPHS, 256, 0, stream>>>(Abuf, starts, r_w1, r_b1, r_w2, r_b2, r_w3, r_b3, out);
}
